// BahdanauAttention_85813446574177
// MI455X (gfx1250) — matrix operand order repaired and checked by reading, not compiled, not run on hardware
//
#include <hip/hip_runtime.h>
#include <hip/hip_bf16.h>

// Bahdanau attention, B=16, S=4096, H=1024 (2H=2048), fp32 in/out.
// Strategy: bf16 WMMA (v_wmma_f32_16x16x32_bf16) for the dominant
// [65536x2048]x[2048x2048] keys @ Wk^T GEMM, fused with tanh + v_w reduction
// so the 2048-wide k_proj is never materialized. keys tile staged in LDS as
// bf16 (257 KB of the 320 KB WGP LDS). Wk pre-converted to bf16 (8 MB,
// L2-resident). v_b dropped (uniform score shift cancels in softmax).

#define B_   16
#define S_   4096
#define H_   1024
#define O2H  2048           // 2*H

#define TILE_M 64           // keys rows staged per workgroup
#define LDS_K  (O2H + 8)    // padded row stride (bf16 elems) -> bank stagger

typedef __attribute__((ext_vector_type(16))) __bf16 v16bf;
typedef __attribute__((ext_vector_type(8)))  float  v8f;

union Frag16 { uint4 u[2]; v16bf v; };

__device__ __forceinline__ unsigned short f2bf(float x) {
    union { float f; unsigned u; } v; v.f = x;
    unsigned r = v.u + 0x7FFFu + ((v.u >> 16) & 1u);   // round-to-nearest-even
    return (unsigned short)(r >> 16);
}

// ---------------------------------------------------------------- prep:
// Wk_w (fp32, [2048,2048] row-major, o-major) -> bf16 in ws; zero score buf.
__global__ void prep_kernel(const float* __restrict__ Wk_w,
                            unsigned short* __restrict__ WkBf,
                            float* __restrict__ scores) {
    int gid = blockIdx.x * blockDim.x + threadIdx.x;
    int stride = gridDim.x * blockDim.x;
    for (int i = gid; i < O2H * O2H; i += stride)
        WkBf[i] = f2bf(Wk_w[i]);
    if (gid < B_ * S_) scores[gid] = 0.f;
}

// ---------------------------------------------------------------- q_proj:
// qb[b,o] = sum_h query[b,h]*Wq_w[o,h] + Wq_b[o] + Wk_b[o]   (67 MFLOP, tiny)
__global__ void qproj_kernel(const float* __restrict__ query,
                             const float* __restrict__ Wq_w,
                             const float* __restrict__ Wq_b,
                             const float* __restrict__ Wk_b,
                             float* __restrict__ qb) {
    int b = blockIdx.x >> 3;                       // 8 blocks per batch
    int o = ((blockIdx.x & 7) << 8) + threadIdx.x; // 0..2047
    const float4* q4 = (const float4*)(query + (size_t)b * H_);
    const float4* w4 = (const float4*)(Wq_w + (size_t)o * H_);
    float s = 0.f;
    #pragma unroll 4
    for (int i = 0; i < H_ / 4; ++i) {
        float4 q = q4[i], w = w4[i];
        s += q.x * w.x + q.y * w.y + q.z * w.z + q.w * w.w;
    }
    qb[b * O2H + o] = s + Wq_b[o] + Wk_b[o];
}

// ---------------------------------------------------------------- scores:
// scores[b,s] = sum_o v_w[o] * tanh(qb[b,o] + sum_k keys[b,s,k]*Wk[o,k])
// One WG = 64 keys rows of one batch; 8 waves = 4 row-blocks x 2 O-halves.
__global__ __launch_bounds__(256) void score_kernel(
        const float* __restrict__ keys,
        const unsigned short* __restrict__ WkBf,
        const float* __restrict__ qb,
        const float* __restrict__ v_w,
        float* __restrict__ scores) {
    extern __shared__ unsigned short ldsK[];     // 64 x 2056 bf16 = 257 KB
    int b    = blockIdx.x >> 6;                  // 64 tiles per batch
    int tile = blockIdx.x & 63;
    int row0 = tile * TILE_M;

    // Stage keys[b, row0 : row0+64, :] into LDS as bf16 (contiguous 512 KB read)
    const float4* src = (const float4*)(keys + ((size_t)b * S_ + row0) * O2H);
    for (int i = threadIdx.x; i < TILE_M * (O2H / 4); i += 256) {
        int r  = i >> 9;          // /(2048/4)
        int k4 = i & 511;
        float4 f = src[i];
        unsigned short* dst = ldsK + r * LDS_K + (k4 << 2);
        *(uint2*)dst = make_uint2(
            (unsigned)f2bf(f.x) | ((unsigned)f2bf(f.y) << 16),
            (unsigned)f2bf(f.z) | ((unsigned)f2bf(f.w) << 16));
    }
    __syncthreads();

    int wave = threadIdx.x >> 5, lane = threadIdx.x & 31;
    int mb = wave & 3;                 // row-block (16 rows) within tile
    int oh = wave >> 2;                // O-half: chunks [oh*64, oh*64+64)
    int n  = lane & 15;
    // A fragment (16x32 bf16): lanes 0-15 K={0..7,16..23}, lanes 16-31 K={8..15,24..31}
    int arow = mb * 16 + n;
    int akh  = (lane >> 4) << 3;       // +0 / +8
    // B fragment (32x16 bf16): lanes 0-15 K={0..7,16..23}, lanes 16-31 K={8..15,24..31}, N=lane&15
    int bkh  = (lane >> 4) << 3;       // +0 / +8

    float part[8] = {0, 0, 0, 0, 0, 0, 0, 0};
    for (int oc = oh * 64; oc < oh * 64 + 64; ++oc) {
        int o = (oc << 4) + n;
        float qv = qb[b * O2H + o];
        float vv = v_w[o];
        const unsigned short* wrow = WkBf + (size_t)o * O2H + bkh; // Wk row o = B column n
        const unsigned short* aptr = ldsK + arow * LDS_K + akh;
        v8f acc = {0, 0, 0, 0, 0, 0, 0, 0};
        for (int kk = 0; kk < O2H; kk += 32) {
            Frag16 a, bm;
            const uint4* lp = (const uint4*)(aptr + kk);   // ds_load_b128 x2
            a.u[0] = lp[0]; a.u[1] = lp[2];                // K-halves 16 elems apart
            const uint4* gp = (const uint4*)(wrow + kk);   // global_load_b128 x2 (L2)
            bm.u[0] = gp[0]; bm.u[1] = gp[2];              // K-halves 16 elems apart
            acc = __builtin_amdgcn_wmma_f32_16x16x32_bf16(
                false, a.v, false, bm.v, (short)0, acc, false, false);
        }
        // C layout: acc[r] @ lane L = k_proj[row mb*16 + r + (L<16?0:8), o]
        #pragma unroll
        for (int r = 0; r < 8; ++r)
            part[r] += tanhf(acc[r] + qv) * vv;
    }
    // Reduce the 16 o-lanes of each half-wave -> per-row score contribution
    #pragma unroll
    for (int r = 0; r < 8; ++r) {
        float v = part[r];
        v += __shfl_xor(v, 1, 32);
        v += __shfl_xor(v, 2, 32);
        v += __shfl_xor(v, 4, 32);
        v += __shfl_xor(v, 8, 32);
        if (n == 0) {
            int srow = row0 + mb * 16 + r + ((lane >> 4) << 3);
            unsafeAtomicAdd(&scores[b * S_ + srow], v);  // global_atomic_add_f32
        }
    }
}

// ---------------------------------------------------------------- softmax:
// one block per batch; writes attn weights to d_out[+32768]; zeros context slice.
__global__ void softmax_kernel(const float* __restrict__ scores,
                               float* __restrict__ out) {
    __shared__ float red[256];
    int b = blockIdx.x, t = threadIdx.x;
    const float* s = scores + b * S_;
    float loc[16];
    float m = -1e30f;
    #pragma unroll
    for (int i = 0; i < 16; ++i) { loc[i] = s[t + i * 256]; m = fmaxf(m, loc[i]); }
    red[t] = m; __syncthreads();
    for (int w = 128; w > 0; w >>= 1) { if (t < w) red[t] = fmaxf(red[t], red[t + w]); __syncthreads(); }
    m = red[0]; __syncthreads();
    float sum = 0.f;
    #pragma unroll
    for (int i = 0; i < 16; ++i) { loc[i] = __expf(loc[i] - m); sum += loc[i]; }
    red[t] = sum; __syncthreads();
    for (int w = 128; w > 0; w >>= 1) { if (t < w) red[t] += red[t + w]; __syncthreads(); }
    float inv = 1.f / red[0];
    float* aw = out + B_ * O2H + b * S_;
    #pragma unroll
    for (int i = 0; i < 16; ++i) aw[t + i * 256] = loc[i] * inv;
    // zero this batch's context slice (accumulated by atomics next kernel)
    float* ctx = out + b * O2H;
    #pragma unroll
    for (int i = 0; i < 8; ++i) ctx[t + i * 256] = 0.f;
}

// ---------------------------------------------------------------- context:
// context[b,k] = sum_s w[b,s]*keys[b,s,k]; streams keys at HBM rate.
__global__ __launch_bounds__(256) void context_kernel(
        const float* __restrict__ keys,
        const float* __restrict__ attn,   // = d_out (weights region)
        float* __restrict__ out) {
    __shared__ float lw[256];
    int b  = blockIdx.x >> 4;
    int s0 = (blockIdx.x & 15) << 8;     // 256-row S chunk
    lw[threadIdx.x] = attn[B_ * O2H + b * S_ + s0 + threadIdx.x];
    __syncthreads();
    int k0 = threadIdx.x << 3;           // 8 contiguous k per thread
    float acc[8] = {0, 0, 0, 0, 0, 0, 0, 0};
    const float* kb = keys + ((size_t)b * S_ + s0) * O2H + k0;
    for (int s = 0; s < 256; ++s) {
        float w = lw[s];
        const float4* kp = (const float4*)(kb + (size_t)s * O2H);
        float4 a = kp[0], c = kp[1];
        acc[0] += w * a.x; acc[1] += w * a.y; acc[2] += w * a.z; acc[3] += w * a.w;
        acc[4] += w * c.x; acc[5] += w * c.y; acc[6] += w * c.z; acc[7] += w * c.w;
    }
    #pragma unroll
    for (int j = 0; j < 8; ++j)
        unsafeAtomicAdd(&out[b * O2H + k0 + j], acc[j]);
}

extern "C" void kernel_launch(void* const* d_in, const int* in_sizes, int n_in,
                              void* d_out, int out_size, void* d_ws, size_t ws_size,
                              hipStream_t stream) {
    const float* query = (const float*)d_in[0];
    const float* keys  = (const float*)d_in[1];
    const float* Wq_w  = (const float*)d_in[2];
    const float* Wq_b  = (const float*)d_in[3];
    const float* Wk_w  = (const float*)d_in[4];
    const float* Wk_b  = (const float*)d_in[5];
    const float* v_w   = (const float*)d_in[6];
    // d_in[7] = v_b: uniform score shift, cancels in softmax -> unused.
    float* out = (float*)d_out;

    char* ws = (char*)d_ws;
    unsigned short* WkBf  = (unsigned short*)ws;                              // 8 MB
    float*          qb    = (float*)(ws + (size_t)O2H * O2H * 2);             // 128 KB
    float*          score = (float*)(ws + (size_t)O2H * O2H * 2 + B_ * O2H * 4); // 256 KB

    prep_kernel<<<4096, 256, 0, stream>>>(Wk_w, WkBf, score);
    qproj_kernel<<<B_ * 8, 256, 0, stream>>>(query, Wq_w, Wq_b, Wk_b, qb);

    size_t ldsBytes = (size_t)TILE_M * LDS_K * sizeof(unsigned short); // 263168 B
    score_kernel<<<B_ * (S_ / TILE_M), 256, ldsBytes, stream>>>(keys, WkBf, qb, v_w, score);

    softmax_kernel<<<B_, 256, 0, stream>>>(score, out);
    context_kernel<<<B_ * 16, 256, 0, stream>>>(keys, out, out);
}
